// SelfAttention_43808666419674
// MI455X (gfx1250) — matrix operand order repaired and checked by reading, not compiled, not run on hardware
//
#include <hip/hip_runtime.h>

// ---------------------------------------------------------------------------
// MI455X (gfx1250) self-attention: bf16 WMMA (f32 accumulate) flash attention.
// Workspace layout (d_ws): Q | K | V as bf16 [B*S, D] = 3 * 32 MB = 96 MB.
// CDNA5 features used: v_wmma_f32_16x16x32_bf16, global_load_async_to_lds_b128
// (ASYNCcnt staging of K tiles), ds_load_b128 (K^T B fragments),
// ds_add_f32 cross-wave score reduction.
// ---------------------------------------------------------------------------

typedef __attribute__((ext_vector_type(16))) __bf16 v16bf;
typedef __attribute__((ext_vector_type(8)))  float  v8f;

union BFrag { v16bf v; unsigned short u[16]; uint4 q[2]; };

__device__ inline unsigned short f2bf(float f) {
  union { float f; unsigned int u; } c; c.f = f;
  unsigned int r = c.u + 0x7FFFu + ((c.u >> 16) & 1u); // round-to-nearest-even
  return (unsigned short)(r >> 16);
}

// K-offset of element e of a bf16 16x32 A-fragment (wave32 layout, ISA 7.12.2):
// lanes 0-15 hold K {0..7,16..23}, lanes 16-31 hold K {8..15,24..31};
// VGPR p holds K pair (base(p), base(p)+1).
__device__ inline int a_koff(int e, int hlf) {
  const int p = e >> 1;
  const int base = (p < 4) ? (2 * p + 8 * hlf) : (8 + 2 * p + 8 * hlf);
  return base + (e & 1);
}

// ---------------------------------------------------------------------------
// GEMM: Y[M,N](bf16) = X[M,K](f32) @ W[K,N](f32) + bias[N]   (M=16384,N=K=1024)
// Block: 256 threads / 8 waves, 128x128 tile; wave -> 32x64 (2x4 WMMA tiles).
// ---------------------------------------------------------------------------
__global__ __launch_bounds__(256) void qkv_gemm_bf16(
    const float* __restrict__ X, const float* __restrict__ W,
    const float* __restrict__ bias, unsigned short* __restrict__ Y,
    int M, int N, int K)
{
  __shared__ unsigned short As[128][32];   // 8 KB
  __shared__ unsigned short Bs[32][128];   // 8 KB

  const int tid  = threadIdx.x;
  const int lane = tid & 31, wave = tid >> 5;
  const int hlf  = lane >> 4, l15 = lane & 15;
  const int m0 = blockIdx.x * 128, n0 = blockIdx.y * 128;
  const int wm = (wave & 3) * 32, wn = (wave >> 2) * 64;

  v8f acc[2][4];
  for (int i = 0; i < 2; ++i)
    for (int j = 0; j < 4; ++j)
      for (int r = 0; r < 8; ++r) acc[i][j][r] = 0.0f;

  for (int k0 = 0; k0 < K; k0 += 32) {
    __syncthreads();
    { // A tile 128x32: thread -> 16 consecutive f32 along k (converted to bf16)
      const int r = tid >> 1, cb = (tid & 1) * 16;
      const float* src = X + (size_t)(m0 + r) * K + k0 + cb;
      #pragma unroll
      for (int c = 0; c < 16; ++c) As[r][cb + c] = f2bf(src[c]);
    }
    { // B tile 32x128: thread -> 16 consecutive f32 along n
      const int r = tid >> 3, cb = (tid & 7) * 16;
      const float* src = W + (size_t)(k0 + r) * N + n0 + cb;
      #pragma unroll
      for (int c = 0; c < 16; ++c) Bs[r][cb + c] = f2bf(src[c]);
    }
    __syncthreads();

    BFrag a[2], b[4];
    #pragma unroll
    for (int i = 0; i < 2; ++i) {
      const int row = wm + i * 16 + l15;
      #pragma unroll
      for (int e = 0; e < 16; ++e) a[i].u[e] = As[row][a_koff(e, hlf)];
    }
    const int bcol = l15;  // B layout: lane <-> N, elements <-> K (table order)
    #pragma unroll
    for (int j = 0; j < 4; ++j) {
      #pragma unroll
      for (int e = 0; e < 16; ++e) b[j].u[e] = Bs[a_koff(e, hlf)][wn + j * 16 + bcol];
    }
    #pragma unroll
    for (int i = 0; i < 2; ++i)
      #pragma unroll
      for (int j = 0; j < 4; ++j)
        acc[i][j] = __builtin_amdgcn_wmma_f32_16x16x32_bf16(
            false, a[i].v, false, b[j].v, (short)0, acc[i][j], false, false);
  }

  // Epilogue: bias add + f32->bf16 store. C layout: VGPR r -> row r + 8*hlf,
  // lane&15 -> col.
  #pragma unroll
  for (int i = 0; i < 2; ++i)
    #pragma unroll
    for (int j = 0; j < 4; ++j) {
      const int col = n0 + wn + j * 16 + l15;
      const float bv = bias[col];
      #pragma unroll
      for (int r = 0; r < 8; ++r) {
        const int row = m0 + wm + i * 16 + r + 8 * hlf;
        Y[(size_t)row * N + col] = f2bf(acc[i][j][r] + bv);
      }
    }
}

// ---------------------------------------------------------------------------
// Flash attention, causal-optional. Block = 16 query rows, 8 waves; wave w owns
// d-slice [128w, 128w+128) for both the QK^T partial sums and the V columns.
// Key tile = 32 keys (so P is a full 16x32 bf16 A-operand, K dim = 32).
// K tile staged into 64 KB dynamic LDS with global_load_async_to_lds_b128.
// V fragments read from global (all of K/V = 64 MB bf16, L2-resident: 192 MB).
// ---------------------------------------------------------------------------
__global__ __launch_bounds__(256) void flash_attn_bf16(
    const unsigned short* __restrict__ Qg, const unsigned short* __restrict__ Kg,
    const unsigned short* __restrict__ Vg, float* __restrict__ Og,
    const int* __restrict__ masking, int S, int D)
{
  extern __shared__ unsigned short Ks[];   // [32][1024] bf16 = 64 KB (dynamic)
  __shared__ float Sc[16][32];             // fp32 score tile
  __shared__ unsigned short Ps[16][32];    // bf16 probabilities
  __shared__ float row_m[16], row_l[16], row_alpha[16];

  const int tid  = threadIdx.x;
  const int lane = tid & 31, wave = tid >> 5;
  const int hlf  = lane >> 4, l15 = lane & 15;
  const int qb = blockIdx.x, bb = blockIdx.y;
  const int q0 = qb * 16;
  const int ds0 = wave * 128;

  // Raw LDS byte address of the dynamic region (starts after static LDS).
  const unsigned ldsK = __builtin_amdgcn_groupstaticsize();

  const unsigned short* Qb = Qg + ((size_t)bb * S + q0) * D;
  const unsigned short* Kb = Kg + (size_t)bb * S * D;
  const unsigned short* Vb = Vg + (size_t)bb * S * D;

  // Preload this wave's Q A-fragments: 4 x (16x32) covering d in [ds0, ds0+128)
  BFrag qf[4];
  #pragma unroll
  for (int t = 0; t < 4; ++t)
    #pragma unroll
    for (int e = 0; e < 16; ++e)
      qf[t].u[e] = Qb[(size_t)l15 * D + ds0 + t * 32 + a_koff(e, hlf)];

  v8f oacc[8];
  for (int j = 0; j < 8; ++j)
    for (int r = 0; r < 8; ++r) oacc[j][r] = 0.0f;

  if (tid < 16) { row_m[tid] = -INFINITY; row_l[tid] = 0.0f; }

  const int cm  = masking[0];
  const int nkt = cm ? ((q0 + 15) / 32 + 1) : (S / 32);

  for (int kt = 0; kt < nkt; ++kt) {
    const int k0 = kt * 32;
    __syncthreads();

    // Stage K tile (32 keys x 1024 d, bf16 = 64 KB) straight into LDS using the
    // CDNA5 async global->LDS path (no VGPR staging, tracked by ASYNCcnt).
    // GVS mode: SGPR 64-bit base + per-lane unsigned byte offset.
    {
      const unsigned short* gbase = Kb + (size_t)k0 * D;
      #pragma unroll
      for (int i = 0; i < 16; ++i) {
        const unsigned boff = (unsigned)(tid + i * 256) * 16u;  // 16B chunks
        const unsigned laddr = ldsK + boff;
        asm volatile("global_load_async_to_lds_b128 %0, %1, %2"
                     :: "v"(laddr), "v"(boff), "s"(gbase) : "memory");
      }
    }
    // Zero the score reduction buffer while the async DMA is in flight.
    ((float*)Sc)[tid] = 0.0f;
    ((float*)Sc)[tid + 256] = 0.0f;
    asm volatile("s_wait_asynccnt 0x0" ::: "memory");
    __syncthreads();

    // Partial scores over this wave's 128-wide d-slice, reduced via ds_add_f32.
    // B operand (32d x 16key) is a transposed access into row-major Ks ->
    // lane's k run is contiguous in its key row, so 2 ds_load_b128 replace 16 ds_load_u16.
    #pragma unroll
    for (int nt = 0; nt < 2; ++nt) {
      v8f sacc;
      for (int r = 0; r < 8; ++r) sacc[r] = 0.0f;
      #pragma unroll
      for (int t = 0; t < 4; ++t) {
        BFrag bk;
        const int d0 = ds0 + t * 32;
        // Per-lane address: lane's 128-bit chunk of the row-major source tile
        // (16 keys x 16 d): row = key0 + l15, col chunk = 8*hlf.
        const unsigned a0 = ldsK +
            (unsigned)(((nt * 16 + l15) * 1024 + d0 + 8 * hlf) * 2);
        const unsigned a1 = a0 + 32u;  // next 16-d sub-tile
        asm volatile("ds_load_b128 %0, %2\n\t"
                     "ds_load_b128 %1, %3"
                     : "=v"(bk.q[0]), "=v"(bk.q[1])
                     : "v"(a0), "v"(a1) : "memory");
        asm volatile("s_wait_dscnt 0x0" ::: "memory");
        sacc = __builtin_amdgcn_wmma_f32_16x16x32_bf16(
            false, qf[t].v, false, bk.v, (short)0, sacc, false, false);
      }
      #pragma unroll
      for (int r = 0; r < 8; ++r)
        atomicAdd(&Sc[r + 8 * hlf][nt * 16 + l15], sacc[r]);
    }
    __syncthreads();

    // Online softmax: wave 0, one row per lane (lanes 0-15)
    if (wave == 0 && lane < 16) {
      const int q = q0 + lane;
      const float m_old = row_m[lane];
      float mx = m_old, sv[32];
      #pragma unroll
      for (int j = 0; j < 32; ++j) {
        float s = Sc[lane][j] * 0.03125f;           // 1/sqrt(1024)
        if (cm && (k0 + j) > q) s = -INFINITY;
        sv[j] = s;
        mx = fmaxf(mx, s);
      }
      const float alpha = __expf(m_old - mx);
      float sum = 0.0f;
      #pragma unroll
      for (int j = 0; j < 32; ++j) {
        const float p = __expf(sv[j] - mx);
        Ps[lane][j] = f2bf(p);
        sum += p;
      }
      row_alpha[lane] = alpha;
      row_m[lane] = mx;
      row_l[lane] = row_l[lane] * alpha + sum;
    }
    __syncthreads();

    // Rescale accumulators, then O += P(16x32) @ V(32 x 128-slice)
    float al[8];
    #pragma unroll
    for (int r = 0; r < 8; ++r) al[r] = row_alpha[r + 8 * hlf];
    #pragma unroll
    for (int j = 0; j < 8; ++j)
      #pragma unroll
      for (int r = 0; r < 8; ++r) oacc[j][r] *= al[r];

    BFrag pf;
    #pragma unroll
    for (int e = 0; e < 16; ++e) pf.u[e] = Ps[l15][a_koff(e, hlf)];

    const unsigned short* vcol = Vb + (size_t)k0 * D + ds0 + l15; // B: lane <-> n
    #pragma unroll
    for (int j = 0; j < 8; ++j) {
      BFrag bv;                                     // elem <-> key (table order)
      #pragma unroll
      for (int e = 0; e < 16; ++e) bv.u[e] = vcol[(size_t)a_koff(e, hlf) * D + j * 16];
      oacc[j] = __builtin_amdgcn_wmma_f32_16x16x32_bf16(
          false, pf.v, false, bv.v, (short)0, oacc[j], false, false);
    }
  }

  __syncthreads();
  float linv[8];
  #pragma unroll
  for (int r = 0; r < 8; ++r) linv[r] = 1.0f / row_l[r + 8 * hlf];
  #pragma unroll
  for (int j = 0; j < 8; ++j)
    #pragma unroll
    for (int r = 0; r < 8; ++r) {
      const int q   = q0 + r + 8 * hlf;
      const int col = ds0 + j * 16 + l15;
      Og[((size_t)bb * S + q) * D + col] = oacc[j][r] * linv[r];
    }
}

// ---------------------------------------------------------------------------
extern "C" void kernel_launch(void* const* d_in, const int* in_sizes, int n_in,
                              void* d_out, int out_size, void* d_ws, size_t ws_size,
                              hipStream_t stream) {
  const float* emb = (const float*)d_in[0];
  const float* Wq  = (const float*)d_in[1];
  const float* bq  = (const float*)d_in[2];
  const float* Wk  = (const float*)d_in[3];
  const float* bk  = (const float*)d_in[4];
  const float* Wv  = (const float*)d_in[5];
  const float* bv  = (const float*)d_in[6];
  const int* masking = (const int*)d_in[7];
  float* out = (float*)d_out;

  const int B = 4, S = 4096, E = 1024, Dh = 1024;
  const int M = B * S;

  unsigned short* Qw = (unsigned short*)d_ws;          // 32 MB
  unsigned short* Kw = Qw + (size_t)M * Dh;            // 32 MB
  unsigned short* Vw = Kw + (size_t)M * Dh;            // 32 MB
  (void)ws_size; (void)in_sizes; (void)n_in; (void)out_size;

  dim3 gblk(M / 128, Dh / 128);                        // 128 x 8 blocks
  qkv_gemm_bf16<<<gblk, 256, 0, stream>>>(emb, Wq, bq, Qw, M, Dh, E);
  qkv_gemm_bf16<<<gblk, 256, 0, stream>>>(emb, Wk, bk, Kw, M, Dh, E);
  qkv_gemm_bf16<<<gblk, 256, 0, stream>>>(emb, Wv, bv, Vw, M, Dh, E);

  dim3 ablk(S / 16, B);                                // 256 x 4 blocks
  const size_t smem = 32 * 1024 * sizeof(unsigned short);  // 64 KB K tile
  flash_attn_bf16<<<ablk, 256, smem, stream>>>(Qw, Kw, Vw, out, masking, S, Dh);
}
